// GreaseArgGraphTransformerLayer_455266533933
// MI455X (gfx1250) — compile-verified, matrix operand order checked by reading, not run on hardware
//
#include <hip/hip_runtime.h>
#include <hip/hip_bf16.h>

#define NN 50000
#define EE 300000
#define DD 256
#define HH 2

typedef __attribute__((ext_vector_type(16))) __bf16 v16bf;
typedef __attribute__((ext_vector_type(8)))  float  v8f;

union FragBF { uint4 u4[2]; v16bf v; };

__device__ __forceinline__ unsigned short f2bf(float f) {
    unsigned u = __float_as_uint(f);
    unsigned r = u + 0x7FFFu + ((u >> 16) & 1u);   // round-to-nearest-even
    return (unsigned short)(r >> 16);
}
__device__ __forceinline__ unsigned fenc(float f) {          // order-preserving uint encode
    unsigned u = __float_as_uint(f);
    return (u & 0x80000000u) ? ~u : (u | 0x80000000u);
}
__device__ __forceinline__ float fdec(unsigned e) {
    unsigned u = (e & 0x80000000u) ? (e ^ 0x80000000u) : ~e;
    return __uint_as_float(u);
}

// ---------------- conversion / packing ----------------
__global__ void k_f32_to_bf16(const float* __restrict__ in, unsigned short* __restrict__ out, int n) {
    int i = blockIdx.x * blockDim.x + threadIdx.x;
    if (i < n) out[i] = f2bf(in[i]);
}

// Wcat[n][k] = W_sel[k][n%256]  (transposed so B-fragment rows are contiguous), n in 0..1023
__global__ void k_pack_wcat(const float* __restrict__ Wq, const float* __restrict__ Wk,
                            const float* __restrict__ Wv, const float* __restrict__ Ws,
                            unsigned short* __restrict__ out) {
    int idx = blockIdx.x * blockDim.x + threadIdx.x;   // 1024*256 threads
    int n = idx >> 8, k = idx & 255;
    int sel = n >> 8, col = n & 255;
    const float* W = (sel == 0) ? Wq : (sel == 1) ? Wk : (sel == 2) ? Wv : Ws;
    out[(size_t)n * 256 + k] = f2bf(W[(size_t)k * 256 + col]);
}

__global__ void k_pack_we(const float* __restrict__ We, unsigned short* __restrict__ out) {
    int idx = blockIdx.x * blockDim.x + threadIdx.x;   // 256*256 threads
    int n = idx >> 8, k = idx & 255;
    out[(size_t)n * 256 + k] = f2bf(We[(size_t)k * 256 + n]);
}

__global__ void k_pack_bias(const float* __restrict__ bq, const float* __restrict__ bk,
                            const float* __restrict__ bv, const float* __restrict__ bs,
                            float* __restrict__ out) {
    int n = blockIdx.x * blockDim.x + threadIdx.x;     // 1024 threads
    int sel = n >> 8, col = n & 255;
    const float* b = (sel == 0) ? bq : (sel == 1) ? bk : (sel == 2) ? bv : bs;
    out[n] = b[col];
}

// ---------------- WMMA bf16 GEMM: out[Mrows x Ncols] = A[Mrows x 256] @ Bt^T (+bias) ----
// Bt stored [Ncols x 256] (transposed). One wave computes a 32(M) x 64(N) tile:
// 2 A fragments x 4 B fragments -> 8 WMMAs per K-step (B reused 2x), with the next
// K-step's fragments prefetched before the WMMA clause. __launch_bounds__(256,1)
// gives regalloc room (~160+ VGPRs) so the prefetch registers stay live.
__global__ __launch_bounds__(256, 1)
void k_gemm_bf16(const unsigned short* __restrict__ A,
                 const unsigned short* __restrict__ Bt,
                 const float* __restrict__ bias,
                 float* __restrict__ outF,
                 unsigned short* __restrict__ outH,
                 int Mtiles32, int Mrows, int Ncols) {
    int wid  = (int)((blockIdx.x * blockDim.x + threadIdx.x) >> 5);
    int lane = threadIdx.x & 31;
    int nt64 = Ncols >> 6;
    int mt = wid / nt64;
    int nt = wid - mt * nt64;
    if (mt >= Mtiles32) return;

    int half = lane >> 4;          // which 16-lane half
    int l16  = lane & 15;
    size_t arow0 = (size_t)(mt * 32 + l16) * 256;
    size_t arow1 = arow0 + (size_t)16 * 256;
    size_t brow[4];
#pragma unroll
    for (int j = 0; j < 4; ++j)
        brow[j] = (size_t)(nt * 64 + j * 16 + l16) * 256;

    v8f acc[2][4] = {};
    FragBF a[2], b[4];
    {
        int kb = half * 8;         // lanes 0-15: K {kb..kb+7, kb+16..kb+23}; lanes 16-31 offset by 8
        a[0].u4[0] = *(const uint4*)(A + arow0 + kb);
        a[0].u4[1] = *(const uint4*)(A + arow0 + kb + 16);
        a[1].u4[0] = *(const uint4*)(A + arow1 + kb);
        a[1].u4[1] = *(const uint4*)(A + arow1 + kb + 16);
#pragma unroll
        for (int j = 0; j < 4; ++j) {
            b[j].u4[0] = *(const uint4*)(Bt + brow[j] + kb);
            b[j].u4[1] = *(const uint4*)(Bt + brow[j] + kb + 16);
        }
    }
#pragma unroll
    for (int ks = 0; ks < 8; ++ks) {
        FragBF an[2], bn[4];
        if (ks < 7) {                              // prefetch next K-step's fragments
            int kn = (ks + 1) * 32 + half * 8;
            an[0].u4[0] = *(const uint4*)(A + arow0 + kn);
            an[0].u4[1] = *(const uint4*)(A + arow0 + kn + 16);
            an[1].u4[0] = *(const uint4*)(A + arow1 + kn);
            an[1].u4[1] = *(const uint4*)(A + arow1 + kn + 16);
#pragma unroll
            for (int j = 0; j < 4; ++j) {
                bn[j].u4[0] = *(const uint4*)(Bt + brow[j] + kn);
                bn[j].u4[1] = *(const uint4*)(Bt + brow[j] + kn + 16);
            }
        }
#pragma unroll
        for (int j = 0; j < 4; ++j) {
            acc[0][j] = __builtin_amdgcn_wmma_f32_16x16x32_bf16(
                false, a[0].v, false, b[j].v, (short)0, acc[0][j], false, false);
            acc[1][j] = __builtin_amdgcn_wmma_f32_16x16x32_bf16(
                false, a[1].v, false, b[j].v, (short)0, acc[1][j], false, false);
        }
        if (ks < 7) {
            a[0] = an[0]; a[1] = an[1];
#pragma unroll
            for (int j = 0; j < 4; ++j) b[j] = bn[j];
        }
    }
#pragma unroll
    for (int i = 0; i < 2; ++i) {
#pragma unroll
        for (int j = 0; j < 4; ++j) {
            int col = nt * 64 + j * 16 + l16;
            float badd = bias ? bias[col] : 0.0f;
#pragma unroll
            for (int r = 0; r < 8; ++r) {
                int row = mt * 32 + i * 16 + half * 8 + r;   // D layout: VGPR r -> M=r / M=8+r per half
                if (row < Mrows) {
                    float val = acc[i][j][r] + badd;
                    if (outF) outF[(size_t)row * Ncols + col] = val;
                    else      outH[(size_t)row * Ncols + col] = f2bf(val);
                }
            }
        }
    }
}

// ---------------- per-edge attention logits + segment max ----------------
__global__ void k_edge_alpha(const int* __restrict__ src, const int* __restrict__ dst,
                             const float* __restrict__ QKVS, const unsigned short* __restrict__ Ep,
                             float* __restrict__ alpha, unsigned* __restrict__ amax) {
    int wid  = (int)((blockIdx.x * blockDim.x + threadIdx.x) >> 5);
    int lane = threadIdx.x & 31;
    if (wid >= EE) return;
    int s = src[wid], d = dst[wid];
    int c = lane * 8;                                   // lanes 0-15 = head0, 16-31 = head1
    const float4* q4 = (const float4*)(QKVS + (size_t)d * 1024 + c);
    const float4* k4 = (const float4*)(QKVS + (size_t)s * 1024 + 256 + c);
    float4 q0 = q4[0], q1 = q4[1];
    float4 k0 = k4[0], k1 = k4[1];
    uint4 eb = *(const uint4*)(Ep + (size_t)wid * 256 + c);
    const unsigned* ew = (const unsigned*)&eb;
    float e[8];
#pragma unroll
    for (int i = 0; i < 4; ++i) {
        e[2 * i]     = __uint_as_float(ew[i] << 16);
        e[2 * i + 1] = __uint_as_float(ew[i] & 0xFFFF0000u);
    }
    float p = q0.x * (k0.x + e[0]) + q0.y * (k0.y + e[1]) + q0.z * (k0.z + e[2]) + q0.w * (k0.w + e[3])
            + q1.x * (k1.x + e[4]) + q1.y * (k1.y + e[5]) + q1.z * (k1.z + e[6]) + q1.w * (k1.w + e[7]);
#pragma unroll
    for (int off = 8; off > 0; off >>= 1) p += __shfl_xor(p, off, 32);   // reduce within 16-lane head group
    if ((lane & 15) == 0) {
        int h = lane >> 4;
        float a = p * 0.08838834764831845f;             // 1/sqrt(128)
        alpha[(size_t)wid * 2 + h] = a;
        atomicMax(&amax[(size_t)d * 2 + h], fenc(a));
    }
}

// ---------------- exp + segment sum ----------------
__global__ void k_edge_exp(const int* __restrict__ dst, float* __restrict__ alpha,
                           const unsigned* __restrict__ amax, float* __restrict__ denom) {
    int i = blockIdx.x * blockDim.x + threadIdx.x;
    if (i >= EE * HH) return;
    int eid = i >> 1, h = i & 1;
    int d = dst[eid];
    float ex = __expf(alpha[i] - fdec(amax[(size_t)d * 2 + h]));
    alpha[i] = ex;
    unsafeAtomicAdd(&denom[(size_t)d * 2 + h], ex);
}

// ---------------- weighted scatter-add aggregation ----------------
__global__ void k_edge_agg(const int* __restrict__ src, const int* __restrict__ dst,
                           const float* __restrict__ QKVS, const unsigned short* __restrict__ Ep,
                           const float* __restrict__ alpha, const float* __restrict__ denom,
                           float* __restrict__ agg) {
    int wid  = (int)((blockIdx.x * blockDim.x + threadIdx.x) >> 5);
    int lane = threadIdx.x & 31;
    if (wid >= EE) return;
    int s = src[wid], d = dst[wid];
    int c = lane * 8;
    int h = lane >> 4;
    float w = alpha[(size_t)wid * 2 + h] / (denom[(size_t)d * 2 + h] + 1e-16f);
    const float4* v4 = (const float4*)(QKVS + (size_t)s * 1024 + 512 + c);
    float4 v0 = v4[0], v1 = v4[1];
    uint4 eb = *(const uint4*)(Ep + (size_t)wid * 256 + c);
    const unsigned* ew = (const unsigned*)&eb;
    float e[8];
#pragma unroll
    for (int i = 0; i < 4; ++i) {
        e[2 * i]     = __uint_as_float(ew[i] << 16);
        e[2 * i + 1] = __uint_as_float(ew[i] & 0xFFFF0000u);
    }
    float* ag = agg + (size_t)d * 256 + c;
    unsafeAtomicAdd(ag + 0, w * (v0.x + e[0]));
    unsafeAtomicAdd(ag + 1, w * (v0.y + e[1]));
    unsafeAtomicAdd(ag + 2, w * (v0.z + e[2]));
    unsafeAtomicAdd(ag + 3, w * (v0.w + e[3]));
    unsafeAtomicAdd(ag + 4, w * (v1.x + e[4]));
    unsafeAtomicAdd(ag + 5, w * (v1.y + e[5]));
    unsafeAtomicAdd(ag + 6, w * (v1.z + e[6]));
    unsafeAtomicAdd(ag + 7, w * (v1.w + e[7]));
}

// ---------------- skip + LayerNorm + ReLU ----------------
__global__ void k_final(const float* __restrict__ agg, const float* __restrict__ QKVS,
                        const float* __restrict__ gamma, const float* __restrict__ beta,
                        float* __restrict__ out) {
    int wid  = (int)((blockIdx.x * blockDim.x + threadIdx.x) >> 5);
    int lane = threadIdx.x & 31;
    if (wid >= NN) return;
    int c = lane * 8;
    const float4* a4 = (const float4*)(agg  + (size_t)wid * 256  + c);
    const float4* s4 = (const float4*)(QKVS + (size_t)wid * 1024 + 768 + c);
    float4 a0 = a4[0], a1 = a4[1], s0 = s4[0], s1 = s4[1];
    float v[8] = { a0.x + s0.x, a0.y + s0.y, a0.z + s0.z, a0.w + s0.w,
                   a1.x + s1.x, a1.y + s1.y, a1.z + s1.z, a1.w + s1.w };
    float sum = 0.f, sq = 0.f;
#pragma unroll
    for (int i = 0; i < 8; ++i) { sum += v[i]; sq += v[i] * v[i]; }
#pragma unroll
    for (int off = 16; off > 0; off >>= 1) {
        sum += __shfl_xor(sum, off, 32);
        sq  += __shfl_xor(sq,  off, 32);
    }
    float mean = sum * (1.0f / 256.0f);
    float var  = sq  * (1.0f / 256.0f) - mean * mean;
    float inv  = rsqrtf(var + 1e-5f);
    const float4* g4 = (const float4*)(gamma + c);
    const float4* b4 = (const float4*)(beta  + c);
    float4 g0 = g4[0], g1 = g4[1], b0 = b4[0], b1 = b4[1];
    float g[8] = { g0.x, g0.y, g0.z, g0.w, g1.x, g1.y, g1.z, g1.w };
    float b[8] = { b0.x, b0.y, b0.z, b0.w, b1.x, b1.y, b1.z, b1.w };
    float* o = out + (size_t)wid * 256 + c;
#pragma unroll
    for (int i = 0; i < 8; ++i)
        o[i] = fmaxf(0.0f, (v[i] - mean) * inv * g[i] + b[i]);
}

extern "C" void kernel_launch(void* const* d_in, const int* in_sizes, int n_in,
                              void* d_out, int out_size, void* d_ws, size_t ws_size,
                              hipStream_t stream) {
    const float* x         = (const float*)d_in[0];
    const float* edge_attr = (const float*)d_in[1];
    const int*   eidx      = (const int*)d_in[2];
    const int*   srcp      = eidx;          // edge_index[0]
    const int*   dstp      = eidx + EE;     // edge_index[1]
    const float* Wq = (const float*)d_in[3],  *bq = (const float*)d_in[4];
    const float* Wk = (const float*)d_in[5],  *bk = (const float*)d_in[6];
    const float* Wv = (const float*)d_in[7],  *bv = (const float*)d_in[8];
    const float* We = (const float*)d_in[9];
    const float* Ws = (const float*)d_in[10], *bs = (const float*)d_in[11];
    const float* gamma = (const float*)d_in[12], *beta = (const float*)d_in[13];
    float* out = (float*)d_out;

    char* ws = (char*)d_ws;
    size_t off = 0;
    auto alloc = [&](size_t bytes) -> void* {
        void* p = ws + off;
        off = (off + bytes + 255) & ~(size_t)255;
        return p;
    };
    unsigned short* xb    = (unsigned short*)alloc((size_t)NN * DD * 2);
    unsigned short* eab   = (unsigned short*)alloc((size_t)EE * DD * 2);
    unsigned short* wcat  = (unsigned short*)alloc((size_t)1024 * 256 * 2);
    unsigned short* web   = (unsigned short*)alloc((size_t)256 * 256 * 2);
    float*          bcat  = (float*)alloc(1024 * 4);
    float*          QKVS  = (float*)alloc((size_t)NN * 1024 * 4);
    unsigned short* Ep    = (unsigned short*)alloc((size_t)EE * DD * 2);
    float*          alpha = (float*)alloc((size_t)EE * HH * 4);
    unsigned*       amax  = (unsigned*)alloc((size_t)NN * HH * 4);
    float*          denom = (float*)alloc((size_t)NN * HH * 4);
    float*          agg   = (float*)alloc((size_t)NN * DD * 4);

    hipMemsetAsync(amax,  0, (size_t)NN * HH * 4, stream);
    hipMemsetAsync(denom, 0, (size_t)NN * HH * 4, stream);
    hipMemsetAsync(agg,   0, (size_t)NN * DD * 4, stream);

    k_f32_to_bf16<<<(NN * DD + 255) / 256, 256, 0, stream>>>(x, xb, NN * DD);
    k_f32_to_bf16<<<(EE * DD + 255) / 256, 256, 0, stream>>>(edge_attr, eab, EE * DD);
    k_pack_wcat<<<1024, 256, 0, stream>>>(Wq, Wk, Wv, Ws, wcat);
    k_pack_we<<<256, 256, 0, stream>>>(We, web);
    k_pack_bias<<<4, 256, 0, stream>>>(bq, bk, bv, bs, bcat);

    // QKVS = x @ [Wq|Wk|Wv|Wskip] + bias : 1563 M32-tiles x 16 N64-tiles = 25008 waves
    k_gemm_bf16<<<3126, 256, 0, stream>>>(xb, wcat, bcat, QKVS, nullptr, 1563, NN, 1024);
    // Eproj = edge_attr @ We (bf16 out)  : 9375 M32-tiles x 4 N64-tiles = 37500 waves
    k_gemm_bf16<<<4688, 256, 0, stream>>>(eab, web, nullptr, nullptr, Ep, 9375, EE, 256);

    k_edge_alpha<<<EE / 8, 256, 0, stream>>>(srcp, dstp, QKVS, Ep, alpha, amax);
    k_edge_exp<<<(EE * HH + 255) / 256, 256, 0, stream>>>(dstp, alpha, amax, denom);
    k_edge_agg<<<EE / 8, 256, 0, stream>>>(srcp, dstp, QKVS, Ep, alpha, denom, agg);
    k_final<<<NN / 8, 256, 0, stream>>>(agg, QKVS, gamma, beta, out);
}
